// SlotAttention_8589934592085
// MI455X (gfx1250) — compile-verified, matrix operand order checked by reading, not run on hardware
//
#include <hip/hip_runtime.h>
#include <hip/hip_bf16.h>
#include <math.h>

typedef __attribute__((ext_vector_type(16))) _Float16 v16h;
typedef __attribute__((ext_vector_type(8)))  float    v8f;

#define BB 16
#define CC 256
#define NPIX 4096
#define KSL 4
#define HID 512
#define NITER 3
#define NSPLIT 8   // K-dim splits for the updates GEMM

// fragment coordinate helper: for (lane, hh) -> k index within a 32-wide K block
__device__ __forceinline__ int frag_k(int lane, int hh) {
    return (hh & 7) + ((hh & 8) ? 16 : 0) + ((lane & 16) ? 8 : 0);
}

// ---------------------------------------------------------------------------
// Kernel 0: pack Wk, Wv (f32, row-major [d][c]) into f16 B-fragments.
// Fragment id = kb*16 + dt  (kb: 32-wide c block, dt: 16-wide d tile),
// element (k=c, n=d):  n = lane&15, k via frag_k.
// ---------------------------------------------------------------------------
__global__ void pack_weights(const float* __restrict__ Wk,
                             const float* __restrict__ Wv,
                             _Float16* __restrict__ wkF,
                             _Float16* __restrict__ wvF) {
    int i = blockIdx.x * 256 + threadIdx.x;         // 2 * 65536
    int mat = i >> 16;
    int r   = i & 65535;
    int fid = r >> 9;                               // kb*16 + dt
    int kb = fid >> 4, dt = fid & 15;
    int le = r & 511;
    int lane = le >> 4, hh = le & 15;
    int c = kb * 32 + frag_k(lane, hh);
    int d = dt * 16 + (lane & 15);
    const float* W = mat ? Wv : Wk;
    _Float16* dst  = mat ? wvF : wkF;
    dst[r] = (_Float16)W[(size_t)d * CC + c];
}

// ---------------------------------------------------------------------------
// Kernel 1: slots = mu + exp(log_sigma) * noise
// ---------------------------------------------------------------------------
__global__ void init_slots(const float* __restrict__ noise,
                           const float* __restrict__ mu,
                           const float* __restrict__ lsig,
                           float* __restrict__ slots) {
    int i = blockIdx.x * 256 + threadIdx.x;         // 16384
    int c = i & (CC - 1);
    slots[i] = mu[c] + __expf(lsig[c]) * noise[i];
}

// ---------------------------------------------------------------------------
// Kernel 2 (fused): LayerNorm over C + k/v projection GEMM via WMMA.
// grid = (256 pixel-tiles, B). 256 threads = 8 waves.
//   k output -> scores-B fragments kB[b][(cb*256+pt)*512 + lane*16 + hh]
//   v output -> updates-B fragments vB[b][(pb*16+ct)*512 + ...], pb = pt/2
// ---------------------------------------------------------------------------
__global__ void proj_kv(const float* __restrict__ x,
                        const float* __restrict__ lnG,
                        const float* __restrict__ lnB,
                        const _Float16* __restrict__ wkF,
                        const _Float16* __restrict__ wvF,
                        const float* __restrict__ bk,
                        const float* __restrict__ bv,
                        _Float16* __restrict__ kB,
                        _Float16* __restrict__ vB) {
    int b  = blockIdx.y;
    int pt = blockIdx.x;
    int t  = threadIdx.x;

    __shared__ float xs[16][CC + 1];
    __shared__ float red[16][16], red2[16][16];
    __shared__ float mv[16][2];
    __shared__ __align__(32) _Float16 aF[8][512];
    __shared__ __align__(32) _Float16 kt[16][CC];
    __shared__ __align__(32) _Float16 vt[16][CC];

    // load x[b, c=t, pt*16 .. pt*16+15]
    const float* xp = x + ((size_t)b * CC + t) * NPIX + pt * 16;
#pragma unroll
    for (int p = 0; p < 16; p++) xs[p][t] = xp[p];
    __syncthreads();

    // per-pixel LN stats: 16 groups of 16 threads
    {
        int gp = t >> 4, j = t & 15;
        float s = 0.f, s2 = 0.f;
        for (int c = j; c < CC; c += 16) { float v = xs[gp][c]; s += v; s2 += v * v; }
        red[gp][j] = s; red2[gp][j] = s2;
    }
    __syncthreads();
    if (t < 16) {
        float m = 0.f, q = 0.f;
#pragma unroll
        for (int i = 0; i < 16; i++) { m += red[t][i]; q += red2[t][i]; }
        m /= CC; q = q / CC - m * m;
        mv[t][0] = m; mv[t][1] = rsqrtf(q + 1e-5f);
    }
    __syncthreads();
    {
        float gg = lnG[t], bb = lnB[t];
#pragma unroll
        for (int p = 0; p < 16; p++)
            xs[p][t] = (xs[p][t] - mv[p][0]) * mv[p][1] * gg + bb;
    }
    __syncthreads();

    // pack A fragments (16 pixels x 32 channels each) into LDS
    {
        int lane = t & 31, kb = t >> 5;
#pragma unroll
        for (int hh = 0; hh < 16; hh++) {
            int k = frag_k(lane, hh);
            aF[kb][lane * 16 + hh] = (_Float16)xs[lane & 15][kb * 32 + k];
        }
    }
    __syncthreads();

    // WMMA: 32 output tiles (16 for k, 16 for v) over 8 waves
    int lane = t & 31, w = t >> 5;
#pragma unroll
    for (int i = 0; i < 4; i++) {
        int tt  = w + 8 * i;
        int mat = tt >> 4, dt = tt & 15;
        const _Float16* wf = (mat ? wvF : wkF) + (size_t)dt * 512;
        v8f acc = {};
#pragma unroll
        for (int kb = 0; kb < 8; kb++) {
            v16h a  = *(const v16h*)&aF[kb][lane * 16];
            v16h bm = *(const v16h*)(wf + (size_t)kb * 16 * 512 + lane * 16);
            acc = __builtin_amdgcn_wmma_f32_16x16x32_f16(false, a, false, bm,
                                                         (short)0, acc, false, false);
        }
        int n = lane & 15;
        int d = dt * 16 + n;
        float bias = mat ? bv[d] : bk[d];
        _Float16* stg = mat ? &vt[0][0] : &kt[0][0];
        int mbase = (lane >= 16) ? 8 : 0;
#pragma unroll
        for (int jj = 0; jj < 8; jj++)
            stg[(mbase + jj) * CC + d] = (_Float16)(acc[jj] + bias);
    }
    __syncthreads();

    // pack k into scores-B fragments (k-dim = channel, n-dim = pixel)
    {
        int cb = w;
        _Float16* dst = kB + (((size_t)b * 8 * 256) + (size_t)cb * 256 + pt) * 512 + lane * 16;
#pragma unroll
        for (int hh = 0; hh < 16; hh++) {
            int k = frag_k(lane, hh);
            dst[hh] = kt[lane & 15][cb * 32 + k];
        }
    }
    // pack v into updates-B fragments (k-dim = pixel, n-dim = channel)
    {
        int pb = pt >> 1;
        int h0 = (pt & 1) ? 8 : 0;
#pragma unroll
        for (int q2 = 0; q2 < 2; q2++) {
            int ct = w * 2 + q2;
            _Float16* dst = vB + (((size_t)b * 128 * 16) + (size_t)pb * 16 + ct) * 512 + lane * 16;
#pragma unroll
            for (int hi = 0; hi < 8; hi++) {
                int hh = h0 + hi;
                int kk = frag_k(lane, hh);
                dst[hh] = vt[kk & 15][ct * 16 + (lane & 15)];
            }
        }
    }
}

// ---------------------------------------------------------------------------
// Kernel 3 (per iter): LN(slots), q = sLN @ Wq^T + bq, pack q A-fragments
// (rows 4..15 zero). grid = B, 256 threads.
// ---------------------------------------------------------------------------
__global__ void slots_q(const float* __restrict__ slots,
                        const float* __restrict__ lnG,
                        const float* __restrict__ lnB,
                        const float* __restrict__ Wq,
                        const float* __restrict__ bq,
                        _Float16* __restrict__ qA) {
    int b = blockIdx.x, t = threadIdx.x;
    __shared__ float sl[KSL][CC], sn[KSL][CC], qs[KSL][CC];
    __shared__ float red[KSL][64], red2[KSL][64], mv2[KSL][2];

    const float* sp = slots + (size_t)b * KSL * CC;
    for (int i = t; i < KSL * CC; i += 256) sl[i >> 8][i & 255] = sp[i];
    __syncthreads();

    {
        int s4 = t >> 6, j = t & 63;
        float a = 0.f, a2 = 0.f;
        for (int c = j; c < CC; c += 64) { float v = sl[s4][c]; a += v; a2 += v * v; }
        red[s4][j] = a; red2[s4][j] = a2;
    }
    __syncthreads();
    if (t < KSL) {
        float m = 0.f, q = 0.f;
#pragma unroll
        for (int i = 0; i < 64; i++) { m += red[t][i]; q += red2[t][i]; }
        m /= CC; q = q / CC - m * m;
        mv2[t][0] = m; mv2[t][1] = rsqrtf(q + 1e-5f);
    }
    __syncthreads();
    for (int i = t; i < KSL * CC; i += 256) {
        int s = i >> 8, c = i & 255;
        sn[s][c] = (sl[s][c] - mv2[s][0]) * mv2[s][1] * lnG[c] + lnB[c];
    }
    __syncthreads();

    // q projection: thread t owns output channel d = t
    {
        const float* wr = Wq + (size_t)t * CC;
        float a0 = bq[t], a1 = bq[t], a2 = bq[t], a3 = bq[t];
        for (int c = 0; c < CC; c++) {
            float wv = wr[c];
            a0 += sn[0][c] * wv; a1 += sn[1][c] * wv;
            a2 += sn[2][c] * wv; a3 += sn[3][c] * wv;
        }
        qs[0][t] = a0; qs[1][t] = a1; qs[2][t] = a2; qs[3][t] = a3;
    }
    __syncthreads();

    // pack q A-fragments (M=16 padded, K=256 -> 8 blocks)
    {
        int lane = t & 31, cb = t >> 5;
        _Float16* dst = qA + ((size_t)b * 8 + cb) * 512 + lane * 16;
        int m = lane & 15;
#pragma unroll
        for (int hh = 0; hh < 16; hh++) {
            int k = frag_k(lane, hh);
            dst[hh] = (m < KSL) ? (_Float16)qs[m][cb * 32 + k] : (_Float16)0.0f;
        }
    }
}

// ---------------------------------------------------------------------------
// Kernel 4 (per iter): WMMA scores, per-pixel softmax over 4 slots,
// pack attn A-fragments, deterministic per-block slot sums.
// grid = (16 pixel groups, B); 8 waves x 2 tiles (tiles even-aligned/wave).
// Per-wave LDS staging => no workgroup barriers needed inside the tile loop.
// ---------------------------------------------------------------------------
__global__ void attn_kernel(const _Float16* __restrict__ qA,
                            const _Float16* __restrict__ kB,
                            _Float16* __restrict__ attnA,
                            float* __restrict__ Spart) {
    int b = blockIdx.y, grp = blockIdx.x;
    int t = threadIdx.x, lane = t & 31, w = t >> 5;
    __shared__ __align__(32) _Float16 att[8][256];   // per-wave 16x16 staging
    __shared__ float wS[8][KSL];

    // zero staging (rows 4..15 stay zero forever)
#pragma unroll
    for (int i = 0; i < 8; i++) att[w][lane * 8 + i] = (_Float16)0.0f;

    // load the 8 q fragments once
    v16h qf[8];
    const _Float16* qb = qA + (size_t)b * 8 * 512;
#pragma unroll
    for (int cb = 0; cb < 8; cb++)
        qf[cb] = *(const v16h*)(qb + (size_t)cb * 512 + lane * 16);

    float ls0 = 0.f, ls1 = 0.f, ls2 = 0.f, ls3 = 0.f;
    const float scale = 0.0625f;                     // C^-0.5
    const _Float16* kbase = kB + (size_t)b * 8 * 256 * 512;

#pragma unroll
    for (int i = 0; i < 2; i++) {
        int pt = grp * 16 + w * 2 + i;
        // prefetch next tile's first fragment while computing this one
        __builtin_prefetch(kbase + ((size_t)(pt + 1)) * 512 + lane * 16, 0, 0);
        v8f acc = {};
#pragma unroll
        for (int cb = 0; cb < 8; cb++) {
            v16h bm = *(const v16h*)(kbase + ((size_t)cb * 256 + pt) * 512 + lane * 16);
            acc = __builtin_amdgcn_wmma_f32_16x16x32_f16(false, qf[cb], false, bm,
                                                         (short)0, acc, false, false);
        }
        // softmax over the 4 real slots (acc[0..3], lanes 0..15)
        float s0 = acc[0] * scale, s1 = acc[1] * scale;
        float s2 = acc[2] * scale, s3 = acc[3] * scale;
        float mx = fmaxf(fmaxf(s0, s1), fmaxf(s2, s3));
        float e0 = __expf(s0 - mx), e1 = __expf(s1 - mx);
        float e2 = __expf(s2 - mx), e3 = __expf(s3 - mx);
        float inv = 1.f / (e0 + e1 + e2 + e3);
        float msk = (lane < 16) ? inv : 0.f;
        float a0 = e0 * msk, a1 = e1 * msk, a2 = e2 * msk, a3 = e3 * msk;
        ls0 += a0; ls1 += a1; ls2 += a2; ls3 += a3;

        // stage rows 0..3 (pixel = lane for lanes<16); same-wave LDS ops are
        // in-order and the compiler inserts s_wait_dscnt for the RAW below.
        if (lane < 16) {
            att[w][0 * 16 + lane] = (_Float16)a0;
            att[w][1 * 16 + lane] = (_Float16)a1;
            att[w][2 * 16 + lane] = (_Float16)a2;
            att[w][3 * 16 + lane] = (_Float16)a3;
        }

        // pack this tile's half of the attn A-fragment (K-dim = pixel)
        int pb = pt >> 1;
        int h0 = (pt & 1) ? 8 : 0;
        _Float16* dst = attnA + ((size_t)b * 128 + pb) * 512 + lane * 16;
#pragma unroll
        for (int hi = 0; hi < 8; hi++) {
            int hh = h0 + hi;
            int kk = frag_k(lane, hh);
            dst[hh] = att[w][(lane & 15) * 16 + (kk & 15)];
        }
    }

    // reduce slot sums across the 16 active lanes (lanes>=16 contribute 0)
#pragma unroll
    for (int off = 8; off; off >>= 1) {
        ls0 += __shfl_xor(ls0, off, 16);
        ls1 += __shfl_xor(ls1, off, 16);
        ls2 += __shfl_xor(ls2, off, 16);
        ls3 += __shfl_xor(ls3, off, 16);
    }
    if (lane == 0) { wS[w][0] = ls0; wS[w][1] = ls1; wS[w][2] = ls2; wS[w][3] = ls3; }
    __syncthreads();
    if (t < KSL) {
        float s = 0.f;
#pragma unroll
        for (int ww = 0; ww < 8; ww++) s += wS[ww][t];
        Spart[((size_t)b * 16 + grp) * KSL + t] = s;
    }
}

// ---------------------------------------------------------------------------
// Kernel 5a (per iter): updates GEMM split over the 4096-pixel K dimension.
// grid = (NSPLIT, B); each block handles 16 pixel-blocks x all 16 ctiles
// (8 waves x 2 ctiles) and writes fp32 partials. 128 blocks -> enough
// parallelism to stream vB (33.5 MB) at full HBM rate.
// ---------------------------------------------------------------------------
__global__ void updates_gemm(const _Float16* __restrict__ attnA,
                             const _Float16* __restrict__ vB,
                             float* __restrict__ updPart) {
    int b = blockIdx.y, sp = blockIdx.x;
    int t = threadIdx.x, lane = t & 31, w = t >> 5;
    const _Float16* ab = attnA + ((size_t)b * 128 + sp * 16) * 512;
    const _Float16* vb = vB + ((size_t)b * 128 * 16 + sp * 16 * 16) * 512;
    float* dst = updPart + ((size_t)b * NSPLIT + sp) * KSL * CC;
#pragma unroll
    for (int q2 = 0; q2 < 2; q2++) {
        int ct = w * 2 + q2;
        v8f acc = {};
        for (int pb = 0; pb < 16; pb++) {
            __builtin_prefetch(vb + ((size_t)(pb + 1) * 16 + ct) * 512 + lane * 16, 0, 0);
            v16h a  = *(const v16h*)(ab + (size_t)pb * 512 + lane * 16);
            v16h bm = *(const v16h*)(vb + ((size_t)pb * 16 + ct) * 512 + lane * 16);
            acc = __builtin_amdgcn_wmma_f32_16x16x32_f16(false, a, false, bm,
                                                         (short)0, acc, false, false);
        }
        if (lane < 16) {
            int c = ct * 16 + lane;
#pragma unroll
            for (int jj = 0; jj < KSL; jj++) dst[jj * CC + c] = acc[jj];
        }
    }
}

// ---------------------------------------------------------------------------
// Kernel 5b (per iter): sum partials, scale by 1/(sumN+eps), then
// GRU-ish update + LN + GELU MLP (fp32 VALU). grid = B, 256 threads.
// ---------------------------------------------------------------------------
__global__ void slot_update(const float* __restrict__ updPart,
                            const float* __restrict__ Spart,
                            float* __restrict__ slots,
                            const float* __restrict__ Wu, const float* __restrict__ bu,
                            const float* __restrict__ lnG, const float* __restrict__ lnB,
                            const float* __restrict__ W1, const float* __restrict__ b1,
                            const float* __restrict__ W2, const float* __restrict__ b2) {
    int b = blockIdx.x, t = threadIdx.x;
    __shared__ float upd[KSL][CC], prev[KSL][CC], snew[KSL][CC], mlp[KSL][CC];
    __shared__ float hbuf[KSL][HID];
    __shared__ float Ssum[KSL];
    __shared__ float red[KSL][64], red2[KSL][64], mv2[KSL][2];

    float* slb = slots + (size_t)b * KSL * CC;
    for (int i = t; i < KSL * CC; i += 256) prev[i >> 8][i & 255] = slb[i];
    if (t < KSL) {
        float s = 0.f;
        const float* sp = Spart + (size_t)b * 16 * KSL;
#pragma unroll
        for (int g2 = 0; g2 < 16; g2++) s += sp[g2 * KSL + t];
        Ssum[t] = 1.f / (s + 1e-8f);
    }
    __syncthreads();

    // reduce the NSPLIT partials and apply 1/(sum+eps)
    {
        const float* up = updPart + (size_t)b * NSPLIT * KSL * CC;
        for (int i = t; i < KSL * CC; i += 256) {
            int s = i >> 8, c = i & 255;
            float acc = 0.f;
#pragma unroll
            for (int sp2 = 0; sp2 < NSPLIT; sp2++)
                acc += up[(size_t)sp2 * KSL * CC + s * CC + c];
            upd[s][c] = acc * Ssum[s];
        }
    }
    __syncthreads();

    // slots = [prev, upd] @ Wu^T + bu + prev   (thread t = output channel d)
    {
        const float* wr = Wu + (size_t)t * (2 * CC);
        float acc[KSL];
#pragma unroll
        for (int s = 0; s < KSL; s++) acc[s] = bu[t] + prev[s][t];
        for (int c = 0; c < CC; c++) {
            float w0 = wr[c], w1 = wr[CC + c];
#pragma unroll
            for (int s = 0; s < KSL; s++) acc[s] += prev[s][c] * w0 + upd[s][c] * w1;
        }
#pragma unroll
        for (int s = 0; s < KSL; s++) snew[s][t] = acc[s];
    }
    __syncthreads();

    // LN(snew) -> mlp
    {
        int s4 = t >> 6, j = t & 63;
        float a1 = 0.f, a2 = 0.f;
        for (int c = j; c < CC; c += 64) { float v = snew[s4][c]; a1 += v; a2 += v * v; }
        red[s4][j] = a1; red2[s4][j] = a2;
    }
    __syncthreads();
    if (t < KSL) {
        float m = 0.f, q = 0.f;
#pragma unroll
        for (int i = 0; i < 64; i++) { m += red[t][i]; q += red2[t][i]; }
        m /= CC; q = q / CC - m * m;
        mv2[t][0] = m; mv2[t][1] = rsqrtf(q + 1e-5f);
    }
    __syncthreads();
    for (int i = t; i < KSL * CC; i += 256) {
        int s = i >> 8, c = i & 255;
        mlp[s][c] = (snew[s][c] - mv2[s][0]) * mv2[s][1] * lnG[c] + lnB[c];
    }
    __syncthreads();

    // h = gelu(mlp @ W1^T + b1)
#pragma unroll
    for (int r = 0; r < 2; r++) {
        int hd = t + r * 256;
        const float* wr = W1 + (size_t)hd * CC;
        float acc[KSL];
#pragma unroll
        for (int s = 0; s < KSL; s++) acc[s] = b1[hd];
        for (int c = 0; c < CC; c++) {
            float wv = wr[c];
#pragma unroll
            for (int s = 0; s < KSL; s++) acc[s] += mlp[s][c] * wv;
        }
#pragma unroll
        for (int s = 0; s < KSL; s++) {
            float xg = acc[s];
            hbuf[s][hd] = xg * 0.5f * (1.f + erff(xg * 0.70710678f));
        }
    }
    __syncthreads();

    // slots_out = snew + h @ W2^T + b2
    {
        const float* wr = W2 + (size_t)t * HID;
        float acc[KSL];
#pragma unroll
        for (int s = 0; s < KSL; s++) acc[s] = snew[s][t] + b2[t];
        for (int c = 0; c < HID; c++) {
            float wv = wr[c];
#pragma unroll
            for (int s = 0; s < KSL; s++) acc[s] += hbuf[s][c] * wv;
        }
#pragma unroll
        for (int s = 0; s < KSL; s++) slb[s * CC + t] = acc[s];
    }
}

// ---------------------------------------------------------------------------
// Kernel 6: readout head. Writes slots then slot_weights into d_out.
// ---------------------------------------------------------------------------
__global__ void head_kernel(const float* __restrict__ slots,
                            const float* __restrict__ We1, const float* __restrict__ be1,
                            const float* __restrict__ We2, const float* __restrict__ be2,
                            float* __restrict__ out) {
    int b = blockIdx.x, t = threadIdx.x;
    __shared__ float sl[KSL][CC], e[KSL][128];
    const float* sp = slots + (size_t)b * KSL * CC;
    for (int i = t; i < KSL * CC; i += 256) {
        float v = sp[i];
        sl[i >> 8][i & 255] = v;
        out[(size_t)b * KSL * CC + i] = v;
    }
    __syncthreads();
    if (t < 128) {
        const float* wr = We1 + (size_t)t * CC;
        float acc[KSL];
#pragma unroll
        for (int s = 0; s < KSL; s++) acc[s] = be1[t];
        for (int c = 0; c < CC; c++) {
            float wv = wr[c];
#pragma unroll
            for (int s = 0; s < KSL; s++) acc[s] += sl[s][c] * wv;
        }
#pragma unroll
        for (int s = 0; s < KSL; s++) {
            float xg = acc[s];
            e[s][t] = xg * 0.5f * (1.f + erff(xg * 0.70710678f));
        }
    }
    __syncthreads();
    if (t < KSL) {
        float a = be2[0];
        for (int c = 0; c < 128; c++) a += e[t][c] * We2[c];
        out[(size_t)BB * KSL * CC + b * KSL + t] = 1.f / (1.f + __expf(-a));
    }
}

// ---------------------------------------------------------------------------
extern "C" void kernel_launch(void* const* d_in, const int* in_sizes, int n_in,
                              void* d_out, int out_size, void* d_ws, size_t ws_size,
                              hipStream_t stream) {
    const float* x       = (const float*)d_in[0];
    const float* noise   = (const float*)d_in[1];
    const float* slot_mu = (const float*)d_in[2];
    const float* slot_ls = (const float*)d_in[3];
    const float* ln_in_g = (const float*)d_in[4];
    const float* ln_in_b = (const float*)d_in[5];
    const float* ln_sl_g = (const float*)d_in[6];
    const float* ln_sl_b = (const float*)d_in[7];
    const float* ln_ml_g = (const float*)d_in[8];
    const float* ln_ml_b = (const float*)d_in[9];
    const float* Wq = (const float*)d_in[10]; const float* bq = (const float*)d_in[11];
    const float* Wk = (const float*)d_in[12]; const float* bk = (const float*)d_in[13];
    const float* Wv = (const float*)d_in[14]; const float* bv = (const float*)d_in[15];
    const float* Wu = (const float*)d_in[16]; const float* bu = (const float*)d_in[17];
    const float* W1 = (const float*)d_in[18]; const float* b1 = (const float*)d_in[19];
    const float* W2 = (const float*)d_in[20]; const float* b2 = (const float*)d_in[21];
    const float* We1 = (const float*)d_in[22]; const float* be1 = (const float*)d_in[23];
    const float* We2 = (const float*)d_in[24]; const float* be2 = (const float*)d_in[25];
    float* out = (float*)d_out;

    // workspace partition (bytes, 256-aligned)
    char* ws = (char*)d_ws;
    size_t off = 0;
    _Float16* wkF  = (_Float16*)(ws + off); off += 131072;                 // 8*16*512 halves
    _Float16* wvF  = (_Float16*)(ws + off); off += 131072;
    _Float16* kB   = (_Float16*)(ws + off); off += (size_t)BB * 8 * 256 * 512 * 2;   // 33.5 MB
    _Float16* vB   = (_Float16*)(ws + off); off += (size_t)BB * 128 * 16 * 512 * 2;  // 33.5 MB
    _Float16* qA   = (_Float16*)(ws + off); off += (size_t)BB * 8 * 512 * 2;
    _Float16* atA  = (_Float16*)(ws + off); off += (size_t)BB * 128 * 512 * 2;
    float*    Sprt = (float*)(ws + off);    off += (size_t)BB * 16 * KSL * 4;
    float*    updP = (float*)(ws + off);    off += (size_t)BB * NSPLIT * KSL * CC * 4;
    float*    slots= (float*)(ws + off);    off += (size_t)BB * KSL * CC * 4;

    pack_weights<<<512, 256, 0, stream>>>(Wk, Wv, wkF, wvF);
    init_slots<<<64, 256, 0, stream>>>(noise, slot_mu, slot_ls, slots);
    proj_kv<<<dim3(256, BB), 256, 0, stream>>>(x, ln_in_g, ln_in_b, wkF, wvF,
                                               bk, bv, kB, vB);
    for (int it = 0; it < NITER; it++) {
        slots_q<<<BB, 256, 0, stream>>>(slots, ln_sl_g, ln_sl_b, Wq, bq, qA);
        attn_kernel<<<dim3(16, BB), 256, 0, stream>>>(qA, kB, atA, Sprt);
        updates_gemm<<<dim3(NSPLIT, BB), 256, 0, stream>>>(atA, vB, updP);
        slot_update<<<BB, 256, 0, stream>>>(updP, Sprt, slots,
                                            Wu, bu, ln_ml_g, ln_ml_b,
                                            W1, b1, W2, b2);
    }
    head_kernel<<<BB, 256, 0, stream>>>(slots, We1, be1, We2, be2, out);
}
